// VectorQuantizer_40492951667444
// MI455X (gfx1250) — compile-verified, matrix operand order checked by reading, not run on hardware
//
#include <hip/hip_runtime.h>
#include <hip/hip_bf16.h>

// ---------------------------------------------------------------------------
// VQ-VAE vector quantizer for MI455X (gfx1250).
// inputs:  [8, 256, 8, 32, 32] f32  (B=8, C=256, M=L*H*W=8192, N=B*M=65536)
// emb:     [1024, 256] f32          (K=1024 codes)
// outputs: quantized [16777216] f32, loss [1] f32, indices [65536] (as f32)
// ---------------------------------------------------------------------------

#define N_ROWS  65536
#define C_DIM   256
#define K_CODES 1024
#define M_SPAT  8192
#define Q_ELEMS 16777216

typedef __attribute__((ext_vector_type(16))) __bf16        v16bf;
typedef __attribute__((ext_vector_type(8)))  float         v8f;
typedef __attribute__((ext_vector_type(4)))  unsigned int  v4u;
typedef __attribute__((ext_vector_type(8)))  int           v8i;
typedef __attribute__((ext_vector_type(4)))  int           v4i;

union FragAB {
    v16bf bf;
    v4u   u[2];
};

__device__ __forceinline__ unsigned short f2bf(float f) {
    unsigned int u = __float_as_uint(f);
    u += 0x7fffu + ((u >> 16) & 1u);           // round-to-nearest-even
    return (unsigned short)(u >> 16);
}

// ---------------------------------------------------------------------------
// Kernel 1: transpose+cast inputs [B][C][M] f32 -> wsX [N][C] bf16
// grid (M/32, C/32, B), block (32, 8)
// ---------------------------------------------------------------------------
__global__ __launch_bounds__(256)
void vq_transpose_cast(const float* __restrict__ x, unsigned short* __restrict__ wsX) {
    __shared__ float tile[32][33];
    const int b  = blockIdx.z;
    const int c0 = blockIdx.y * 32;
    const int m0 = blockIdx.x * 32;
    const int tx = threadIdx.x, ty = threadIdx.y;
#pragma unroll
    for (int i = 0; i < 4; ++i) {
        int c = c0 + ty + i * 8;
        tile[ty + i * 8][tx] = x[((size_t)b * C_DIM + c) * M_SPAT + m0 + tx];
    }
    __syncthreads();
#pragma unroll
    for (int i = 0; i < 4; ++i) {
        int m = m0 + ty + i * 8;
        wsX[((size_t)b * M_SPAT + m) * C_DIM + c0 + tx] = f2bf(tile[tx][ty + i * 8]);
    }
}

// ---------------------------------------------------------------------------
// Kernel 2: codebook prep: bf16 cast + ||e_k||^2, zero loss slot.
// grid (1024), block (256) -- one code row per block
// ---------------------------------------------------------------------------
__global__ __launch_bounds__(256)
void vq_prep_codebook(const float* __restrict__ emb, unsigned short* __restrict__ wsE,
                      float* __restrict__ enorm, float* __restrict__ outLoss) {
    __shared__ float red[256];
    const int k = blockIdx.x, c = threadIdx.x;
    float e = emb[(size_t)k * C_DIM + c];
    wsE[(size_t)k * C_DIM + c] = f2bf(e);
    red[c] = e * e;
    __syncthreads();
    for (int s = 128; s > 0; s >>= 1) {
        if (c < s) red[c] += red[c + s];
        __syncthreads();
    }
    if (c == 0) enorm[k] = red[0];
    if (k == 0 && c == 0) *outLoss = 0.f;
}

// ---------------------------------------------------------------------------
// Kernel 3: fused GEMM (bf16 WMMA) + argmin.
// Block = 256 threads (8 waves) handles 128 rows x 1024 codes.
// A tile (128x256 bf16 = 64KB) pulled into LDS via TDM tensor_load_to_lds.
// Each wave: 16 rows x 128 codes = 8 accumulator tiles of 16x16 f32.
// dist = ||e||^2 - 2*x.e  (||x||^2 dropped: constant per row for argmin)
// ---------------------------------------------------------------------------
__global__ __launch_bounds__(256)
void vq_argmin_wmma(const unsigned short* __restrict__ wsX,
                    const unsigned short* __restrict__ wsE,
                    const float* __restrict__ enorm,
                    int* __restrict__ wsIdx, float* __restrict__ outIdxF) {
    __shared__ unsigned short tileA[128 * C_DIM];   // 64KB
    const int n0 = blockIdx.x * 128;

    // ---- TDM: DMA the 128x256 bf16 A-tile from global into LDS -----------
    if (threadIdx.x == 0) {
        unsigned long long ga = (unsigned long long)(wsX + (size_t)n0 * C_DIM);
        unsigned int la = (unsigned int)(unsigned long long)&tileA[0];
        // D# group0: count=1 | lds_addr | global_addr[56:0] | type=2
        v4u g0 = { 1u, la,
                   (unsigned int)(ga & 0xffffffffull),
                   (unsigned int)((ga >> 32) & 0x1ffffffull) | (2u << 30) };
        // D# group1: data_size=2B; tensor_dim0=256, tensor_dim1=65536;
        //            tile_dim0=256, tile_dim1=128; dim0_stride=256
        v8i g1 = { (int)(1u << 16),            // data_size=1 (2 bytes), mask=0
                   (int)(256u << 16),          // tensor_dim0[15:0] << 16
                   0,                          // dim0[31:16]=0 | dim1[15:0]=0
                   (int)(1u | (256u << 16)),   // dim1[31:16]=1 | tile_dim0=256
                   128,                        // tile_dim1=128, tile_dim2=0
                   256,                        // tensor_dim0_stride[31:0]
                   0, 0 };
        v4i gz = { 0, 0, 0, 0 };
#if defined(__clang_major__) && (__clang_major__ >= 23)
        v8i gz8 = { 0, 0, 0, 0, 0, 0, 0, 0 };
        __builtin_amdgcn_tensor_load_to_lds(g0, g1, gz, gz, gz8, 0);
#else
        __builtin_amdgcn_tensor_load_to_lds(g0, g1, gz, gz, 0);
#endif
    }
    __builtin_amdgcn_s_wait_tensorcnt(0);
    __syncthreads();

    const int lane = threadIdx.x & 31;
    const int wv   = threadIdx.x >> 5;
    const int ml   = lane & 15;      // M index (A) / N column (B,C,D)
    const int hi   = lane >> 4;      // half-wave selector
    const int rowA = wv * 16 + ml;

    float best[8];
    int   bestIdx[8];
#pragma unroll
    for (int j = 0; j < 8; ++j) { best[j] = 3.4e38f; bestIdx[j] = 0; }

    const v8f vzero = { 0.f, 0.f, 0.f, 0.f, 0.f, 0.f, 0.f, 0.f };

    for (int kc = 0; kc < K_CODES; kc += 128) {
        v8f acc[8];
#pragma unroll
        for (int t = 0; t < 8; ++t) acc[t] = vzero;

        for (int cc = 0; cc < C_DIM; cc += 32) {
            // A fragment 16x32 bf16 from LDS (per-lane: K hi*8..+7, 16+hi*8..+7)
            FragAB aF;
            aF.u[0] = *(const v4u*)&tileA[rowA * C_DIM + cc + hi * 8];
            aF.u[1] = *(const v4u*)&tileA[rowA * C_DIM + cc + 16 + hi * 8];
#pragma unroll
            for (int t = 0; t < 8; ++t) {
                // B fragment 32x16 bf16: column = code row (L2-hot, 512KB)
                size_t eb = ((size_t)(kc + t * 16 + ml)) * C_DIM + cc + hi * 8;
                FragAB bF;
                bF.u[0] = *(const v4u*)&wsE[eb];
                bF.u[1] = *(const v4u*)&wsE[eb + 16];
                acc[t] = __builtin_amdgcn_wmma_f32_16x16x32_bf16(
                    false, aF.bf, false, bF.bf, (short)0, acc[t], false, false);
            }
        }
        // epilogue: dist = ||e||^2 - 2*acc ; running per-row argmin
#pragma unroll
        for (int t = 0; t < 8; ++t) {
            int n = kc + t * 16 + ml;
            float en = enorm[n];
#pragma unroll
            for (int j = 0; j < 8; ++j) {
                float d = fmaf(-2.f, acc[t][j], en);
                if (d < best[j]) { best[j] = d; bestIdx[j] = n; }
            }
        }
    }

    // min-reduce across the 16 lanes sharing each row (lanes 0-15 / 16-31)
#pragma unroll
    for (int j = 0; j < 8; ++j) {
        float v = best[j];
        int   bi = bestIdx[j];
#pragma unroll
        for (int mask = 8; mask >= 1; mask >>= 1) {
            float ov = __shfl_xor(v, mask, 32);
            int   oi = __shfl_xor(bi, mask, 32);
            if (ov < v || (ov == v && oi < bi)) { v = ov; bi = oi; }
        }
        if (ml == 0) {
            int row = n0 + wv * 16 + hi * 8 + j;
            wsIdx[row]   = bi;
            outIdxF[row] = (float)bi;
        }
    }
}

// ---------------------------------------------------------------------------
// Kernel 4: gather fp32 codebook rows to [B,C,M] output + loss reduction.
// loss = 1.25 * mean((q - x)^2)    (q_latent + 0.25*e_latent, forward value)
// grid (N/256), block (256): thread <-> position n; loop over c (coalesced m)
// ---------------------------------------------------------------------------
__global__ __launch_bounds__(256)
void vq_gather_loss(const int* __restrict__ wsIdx, const float* __restrict__ emb,
                    const float* __restrict__ x, float* __restrict__ outQ,
                    float* __restrict__ outLoss) {
    __shared__ float red[256];
    const int n  = blockIdx.x * 256 + threadIdx.x;
    const int b  = n >> 13;           // n / 8192
    const int m  = n & (M_SPAT - 1);
    const int id = wsIdx[n];
    const float* er = emb + (size_t)id * C_DIM;
    const size_t base = (size_t)b * C_DIM * M_SPAT + m;
    float s = 0.f;
#pragma unroll 4
    for (int c = 0; c < C_DIM; ++c) {
        float q = er[c];
        size_t off = base + (size_t)c * M_SPAT;
        float xv = x[off];
        outQ[off] = q;
        float d = q - xv;
        s = fmaf(d, d, s);
    }
    red[threadIdx.x] = s;
    __syncthreads();
    for (int st = 128; st > 0; st >>= 1) {
        if (threadIdx.x < st) red[threadIdx.x] += red[threadIdx.x + st];
        __syncthreads();
    }
    if (threadIdx.x == 0)
        atomicAdd(outLoss, red[0] * (1.25f / (float)Q_ELEMS));
}

// ---------------------------------------------------------------------------
extern "C" void kernel_launch(void* const* d_in, const int* in_sizes, int n_in,
                              void* d_out, int out_size, void* d_ws, size_t ws_size,
                              hipStream_t stream) {
    (void)in_sizes; (void)n_in; (void)out_size; (void)ws_size;
    const float* x   = (const float*)d_in[0];
    const float* emb = (const float*)d_in[1];

    float* outQ    = (float*)d_out;            // [16777216]
    float* outLoss = outQ + Q_ELEMS;           // [1]
    float* outIdxF = outQ + Q_ELEMS + 1;       // [65536]

    char* ws = (char*)d_ws;
    unsigned short* wsX   = (unsigned short*)ws;                          // 32 MB
    unsigned short* wsE   = (unsigned short*)(ws + 33554432);             // 512 KB
    float*          enorm = (float*)(ws + 33554432 + 524288);             // 4 KB
    int*            wsIdx = (int*)(ws + 33554432 + 524288 + 4096);        // 256 KB

    vq_transpose_cast<<<dim3(M_SPAT / 32, C_DIM / 32, 8), dim3(32, 8), 0, stream>>>(x, wsX);
    vq_prep_codebook<<<K_CODES, 256, 0, stream>>>(emb, wsE, enorm, outLoss);
    vq_argmin_wmma<<<N_ROWS / 128, 256, 0, stream>>>(wsX, wsE, enorm, wsIdx, outIdxF);
    vq_gather_loss<<<N_ROWS / 256, 256, 0, stream>>>(wsIdx, emb, x, outQ, outLoss);
}
